// Transformer_76553497084574
// MI455X (gfx1250) — matrix operand order repaired and checked by reading, not compiled, not run on hardware
//
#include <hip/hip_runtime.h>
#include <hip/hip_bf16.h>
#include <math.h>

// ---------------- types ----------------
typedef _Float16 half16v __attribute__((ext_vector_type(16)));   // 32B: one WMMA operand per lane
typedef float    float8v __attribute__((ext_vector_type(8)));

union FloatVec { float8v v; float e[8]; };
union HalfVec  { half16v v; _Float16 e[16]; };

static __device__ inline float8v wmma_f16f32(half16v a, half16v b, float8v c) {
  // D = A(16x32 f16) * B(32x16 f16) + C(16x16 f32)
  return __builtin_amdgcn_wmma_f32_16x16x32_f16(false, a, false, b, (short)0, c, false, false);
}

static __device__ inline half16v lds_read_frag(const _Float16* p) {
  return *(const half16v*)p;   // 32B aligned -> 2x ds_load_b128
}

// CDNA5 async memory->LDS copy (16B per lane), tracked by ASYNCcnt.
// Generic LDS pointer: low 32 bits are the DS byte address (ISA 10.2).
static __device__ inline void async_copy_b128(const _Float16* gsrc, _Float16* lds_dst) {
  unsigned lds = (unsigned)(unsigned long long)lds_dst;
  asm volatile("global_load_async_to_lds_b128 %0, %1, off"
               :: "v"(lds), "v"(gsrc) : "memory");
}
static __device__ inline void async_wait0() {
  asm volatile("s_wait_asynccnt 0x0" ::: "memory");
}

// Fragment-layout LDS: frag[lane][i], lane row = 16 halves = 32B contiguous.
// A-layout inverse (16x32 f16):  (row,k) -> lane=(row&15)+16*((k>>3)&1), i=(k&7)+8*(k>>4)
// B-layout inverse (32x16 f16):  (k,n)   -> lane=(n&15)+16*((k>>3)&1), i=(k&7)+8*(k>>4)

// ---------------- conversion ----------------
__global__ void cvt_f32_to_f16_kernel(const float* __restrict__ src,
                                      _Float16* __restrict__ dst, size_t n) {
  size_t i = (size_t)blockIdx.x * blockDim.x + threadIdx.x;
  size_t stride = (size_t)gridDim.x * blockDim.x;
  for (; i < n; i += stride) dst[i] = (_Float16)src[i];
}

// ---------------- layernorm (f32 in -> f16 out) ----------------
__global__ void ln_kernel(const float* __restrict__ x, const float* __restrict__ g,
                          const float* __restrict__ b, _Float16* __restrict__ out, int C) {
  __shared__ float red[256], red2[256];
  const int row = blockIdx.x;
  const int tid = threadIdx.x;
  const float* xr = x + (size_t)row * C;
  float s = 0.f, s2 = 0.f;
  for (int c = tid; c < C; c += 256) { float v = xr[c]; s += v; s2 += v * v; }
  red[tid] = s; red2[tid] = s2;
  __syncthreads();
  for (int off = 128; off > 0; off >>= 1) {
    if (tid < off) { red[tid] += red[tid + off]; red2[tid] += red2[tid + off]; }
    __syncthreads();
  }
  const float mu = red[0] / C;
  const float var = red2[0] / C - mu * mu;
  const float rstd = rsqrtf(var + 1e-5f);
  _Float16* orow = out + (size_t)row * C;
  for (int c = tid; c < C; c += 256)
    orow[c] = (_Float16)((xr[c] - mu) * rstd * g[c] + b[c]);
}

// ---------------- generic WMMA GEMM (128x64 tile, double-buffered async) ----------------
// out[m,n] = sum_k A[m,k] * Wrow[n,k] + bias[n]   (Wrow is [N,K] row-major)
#define GEMM_F32_RES  0
#define GEMM_F16      1
#define GEMM_F16_GELU 2

__global__ __launch_bounds__(128)
void gemm_wmma_kernel(const _Float16* __restrict__ A, const _Float16* __restrict__ Wrow,
                      const float* __restrict__ bias, const float* __restrict__ resid,
                      float* __restrict__ outF, _Float16* __restrict__ outH,
                      int M, int N, int K, int mode) {
  // double-buffered fragment-layout staging
  __shared__ alignas(32) _Float16 Af[2][8][32][16];  // [buf][row-subtile (16 rows)][lane][i]
  __shared__ alignas(32) _Float16 Bf[2][4][32][16];  // [buf][col-subtile][lane][i]
  const int tid  = threadIdx.x;
  const int lane = tid & 31;
  const int wave = tid >> 5;
  const int n0 = blockIdx.x * 64;
  const int m0 = blockIdx.y * 128;
  const int bn = tid >> 1, bkh = tid & 1;   // B staging coords

  FloatVec acc[2][4];
#pragma unroll
  for (int a = 0; a < 2; ++a)
#pragma unroll
    for (int j = 0; j < 4; ++j) acc[a][j].v = (float8v){};

  auto stage = [&](int kc, int buf) {
    // A: 128 rows x 4 k-octets, 16B each; exactly 4 transfers/thread (no guards)
#pragma unroll
    for (int u = 0; u < 4; ++u) {
      int e = tid + u * 128;
      int row = e >> 2, j = e & 3;
      async_copy_b128(&A[(size_t)(m0 + row) * K + kc + j * 8],
                      &Af[buf][row >> 4][(row & 15) + 16 * (j & 1)][8 * (j >> 1)]);
    }
    // B: 64 cols x 2 k-halves, 32B each; 1 transfer pair/thread
    const _Float16* s = &Wrow[(size_t)(n0 + bn) * K + kc + bkh * 8];
    _Float16* d = &Bf[buf][bn >> 4][(bn & 15) + 16 * bkh][0];
    async_copy_b128(s, d);
    async_copy_b128(s + 16, d + 8);
  };

  stage(0, 0);
  async_wait0();
  __syncthreads();

  const int nk = K >> 5;
  for (int t = 0; t < nk; ++t) {
    const int cur = t & 1;
    if (t + 1 < nk) stage((t + 1) * 32, cur ^ 1);

    half16v a0 = lds_read_frag(&Af[cur][2 * wave][lane][0]);
    half16v a1 = lds_read_frag(&Af[cur][2 * wave + 1][lane][0]);
#pragma unroll
    for (int j = 0; j < 4; ++j) {
      half16v bB = lds_read_frag(&Bf[cur][j][lane][0]);
      acc[0][j].v = wmma_f16f32(a0, bB, acc[0][j].v);
      acc[1][j].v = wmma_f16f32(a1, bB, acc[1][j].v);
    }
    async_wait0();
    __syncthreads();
  }

  // epilogue — C layout: row = r + 8*(lane>>4), col = lane&15
  const int g = lane >> 4;
#pragma unroll
  for (int a = 0; a < 2; ++a) {
#pragma unroll
    for (int j = 0; j < 4; ++j) {
#pragma unroll
      for (int r = 0; r < 8; ++r) {
        int m = m0 + (2 * wave + a) * 16 + r + 8 * g;
        int n = n0 + j * 16 + (lane & 15);
        float v = acc[a][j].e[r] + bias[n];
        if (mode == GEMM_F32_RES) {
          outF[(size_t)m * N + n] = v + resid[(size_t)m * N + n];
        } else if (mode == GEMM_F16) {
          outH[(size_t)m * N + n] = (_Float16)v;
        } else {
          float ge = 0.5f * v * (1.0f + erff(v * 0.70710678118654752f));
          outH[(size_t)m * N + n] = (_Float16)ge;
        }
      }
    }
  }
}

// ---------------- flash attention ----------------
// qkv: [B*N, 3*C] f16 (q | k | v, each [H, 64]); out o16: [B*N, C] f16
__global__ __launch_bounds__(128)
void attn_kernel(const _Float16* __restrict__ qkv, _Float16* __restrict__ o16,
                 int Ntok, int C3, int Cc, float scale) {
  // all tiles kept directly in fragment layout
  __shared__ alignas(32) _Float16 Qf[4][2][32][16];  // [wave][k-half(hd)][lane][i]
  __shared__ alignas(32) _Float16 Kf[4][2][32][16];  // [key-subtile][hd-half][lane][i]
  __shared__ alignas(32) _Float16 Vf[4][2][32][16];  // [hd-subtile][key-half][lane][i]
  __shared__ alignas(32) _Float16 Pf[4][2][32][16];  // [wave][key-half][lane][i]
  const int tid  = threadIdx.x;
  const int lane = tid & 31;
  const int wave = tid >> 5;
  const int rt = blockIdx.x;   // query row tile (64 rows)
  const int h  = blockIdx.y;   // head
  const int b  = blockIdx.z;   // batch
  const int g  = lane >> 4;

  // ---- stage Q into A-fragment layout: 64 rows x 8 octets, async 16B each ----
#pragma unroll
  for (int u = 0; u < 4; ++u) {
    int e = tid + u * 128;
    int row = e >> 3, oct = e & 7;                 // d = oct*8 .. +7
    int tok = b * Ntok + rt * 64 + row;
    int kh = oct >> 2, j2 = oct & 3;
    async_copy_b128(&qkv[(size_t)tok * C3 + h * 64 + oct * 8],
                    &Qf[row >> 4][kh][(row & 15) + 16 * (j2 & 1)][8 * (j2 >> 1)]);
  }
  async_wait0();
  __syncthreads();
  half16v aQ0 = lds_read_frag(&Qf[wave][0][lane][0]);
  half16v aQ1 = lds_read_frag(&Qf[wave][1][lane][0]);

  float m_i[8], l_i[8];
  FloatVec o_acc[4];
#pragma unroll
  for (int r = 0; r < 8; ++r) { m_i[r] = -3.0e38f; l_i[r] = 0.f; }
#pragma unroll
  for (int j = 0; j < 4; ++j) o_acc[j].v = (float8v){};

  for (int c = 0; c < 16; ++c) {                   // 16 chunks of 64 keys
    __syncthreads();                               // protect K/V/P reuse
    // ---- stage K^T fragments async: per (key, hd-16-group t): 32B each ----
#pragma unroll
    for (int u = 0; u < 2; ++u) {
      int e = tid + u * 128;
      int key = e >> 2, t = e & 3;                 // hd = 32*(t>>1) + 8*(t&1) + {0..7, 16..23}
      int tok = b * Ntok + c * 64 + key;
      const _Float16* s = &qkv[(size_t)tok * C3 + Cc + h * 64 + (t >> 1) * 32 + (t & 1) * 8];
      _Float16* d = &Kf[key >> 4][t >> 1][(key & 15) + 16 * (t & 1)][0];
      async_copy_b128(s, d);
      async_copy_b128(s + 16, d + 8);
    }
    // ---- stage V fragments: per (hd, key-quarter q): gather 16 keys ----
#pragma unroll
    for (int u = 0; u < 2; ++u) {
      int e = tid + u * 128;
      int hd = e >> 2, q = e & 3;                  // keys 32*(q>>1) + 8*(q&1) + {0..7, 16..23}
      HalfVec tmp;
#pragma unroll
      for (int i = 0; i < 16; ++i) {
        int tok = b * Ntok + c * 64 + (q >> 1) * 32 + (q & 1) * 8 + (i & 7) + 16 * (i >> 3);
        tmp.e[i] = qkv[(size_t)tok * C3 + 2 * Cc + h * 64 + hd];
      }
      *(half16v*)&Vf[hd >> 4][q >> 1][(hd & 15) + 16 * (q & 1)][0] = tmp.v;
    }
    async_wait0();
    __syncthreads();

    // ---- S = Q * K^T (contraction over HD=64) ----
    FloatVec s[4];
#pragma unroll
    for (int j = 0; j < 4; ++j) {
      s[j].v = (float8v){};
      s[j].v = wmma_f16f32(aQ0, lds_read_frag(&Kf[j][0][lane][0]), s[j].v);
      s[j].v = wmma_f16f32(aQ1, lds_read_frag(&Kf[j][1][lane][0]), s[j].v);
    }

    // ---- online softmax, per row r (row = r + 8*g) ----
#pragma unroll
    for (int r = 0; r < 8; ++r) {
      float mx = fmaxf(fmaxf(s[0].e[r] * scale, s[1].e[r] * scale),
                       fmaxf(s[2].e[r] * scale, s[3].e[r] * scale));
#pragma unroll
      for (int off = 8; off > 0; off >>= 1) mx = fmaxf(mx, __shfl_xor(mx, off, 16));
      float mnew = fmaxf(m_i[r], mx);
      float alpha = __expf(m_i[r] - mnew);
      float rs = 0.f;
#pragma unroll
      for (int j = 0; j < 4; ++j) {
        float pe = __expf(s[j].e[r] * scale - mnew);
        s[j].e[r] = pe;
        rs += pe;
      }
#pragma unroll
      for (int off = 8; off > 0; off >>= 1) rs += __shfl_xor(rs, off, 16);
      l_i[r] = l_i[r] * alpha + rs;
      m_i[r] = mnew;
#pragma unroll
      for (int j = 0; j < 4; ++j) o_acc[j].e[r] *= alpha;
    }

    // ---- spill P (C layout -> A fragment layout) ----
#pragma unroll
    for (int j = 0; j < 4; ++j)
#pragma unroll
      for (int r = 0; r < 8; ++r)
        Pf[wave][j >> 1][(r + 8 * g) + 16 * ((lane >> 3) & 1)][(lane & 7) + 8 * (j & 1)] =
            (_Float16)s[j].e[r];
    __syncthreads();

    half16v aP0 = lds_read_frag(&Pf[wave][0][lane][0]);
    half16v aP1 = lds_read_frag(&Pf[wave][1][lane][0]);
#pragma unroll
    for (int j = 0; j < 4; ++j) {
      o_acc[j].v = wmma_f16f32(aP0, lds_read_frag(&Vf[j][0][lane][0]), o_acc[j].v);
      o_acc[j].v = wmma_f16f32(aP1, lds_read_frag(&Vf[j][1][lane][0]), o_acc[j].v);
    }
  }

  // ---- write O (normalize by l) ----
#pragma unroll
  for (int j = 0; j < 4; ++j)
#pragma unroll
    for (int r = 0; r < 8; ++r) {
      int tok = b * Ntok + rt * 64 + wave * 16 + r + 8 * g;
      int col = h * 64 + j * 16 + (lane & 15);
      o16[(size_t)tok * Cc + col] = (_Float16)(o_acc[j].e[r] / l_i[r]);
    }
}

// ---------------- host-side launch ----------------
extern "C" void kernel_launch(void* const* d_in, const int* in_sizes, int n_in,
                              void* d_out, int out_size, void* d_ws, size_t ws_size,
                              hipStream_t stream) {
  (void)in_sizes; (void)n_in; (void)out_size; (void)ws_size;
  constexpr int Bb = 4, Nn = 1024, Cc = 768, Ll = 4, Hh = 12, MLPn = 3072;
  constexpr int M = Bb * Nn;          // 4096 tokens
  constexpr int C3 = 3 * Cc;          // 2304
  const float scale = 0.125f;         // 64^-0.5

  const float* x     = (const float*)d_in[0];
  const float* Wqkv  = (const float*)d_in[1];
  const float* bqkv  = (const float*)d_in[2];
  const float* Wproj = (const float*)d_in[3];
  const float* bproj = (const float*)d_in[4];
  const float* ln1g  = (const float*)d_in[5];
  const float* ln1b  = (const float*)d_in[6];
  const float* W1    = (const float*)d_in[7];
  const float* b1    = (const float*)d_in[8];
  const float* W2    = (const float*)d_in[9];
  const float* b2    = (const float*)d_in[10];
  const float* ln2g  = (const float*)d_in[11];
  const float* ln2b  = (const float*)d_in[12];

  // ---- workspace carve-out ----
  char* p = (char*)d_ws;
  auto carve = [&](size_t bytes) -> void* {
    void* r = (void*)p;
    p += (bytes + 255) & ~(size_t)255;
    return r;
  };
  const size_t nWqkv  = (size_t)Ll * C3 * Cc;
  const size_t nWproj = (size_t)Ll * Cc * Cc;
  const size_t nW1    = (size_t)Ll * MLPn * Cc;
  const size_t nW2    = (size_t)Ll * Cc * MLPn;
  _Float16* wqkv16  = (_Float16*)carve(nWqkv  * 2);
  _Float16* wproj16 = (_Float16*)carve(nWproj * 2);
  _Float16* w116    = (_Float16*)carve(nW1    * 2);
  _Float16* w216    = (_Float16*)carve(nW2    * 2);
  _Float16* h16     = (_Float16*)carve((size_t)M * Cc   * 2);
  _Float16* qkv16   = (_Float16*)carve((size_t)M * C3   * 2);
  _Float16* o16     = (_Float16*)carve((size_t)M * Cc   * 2);
  _Float16* ff16    = (_Float16*)carve((size_t)M * MLPn * 2);
  float*    bufM    = (float*)carve((size_t)M * Cc * 4);
  float*    bufA    = (float*)carve((size_t)M * Cc * 4);
  float*    bufB    = (float*)carve((size_t)M * Cc * 4);

  // ---- one-time weight conversion f32 -> f16 ----
  cvt_f32_to_f16_kernel<<<2048, 256, 0, stream>>>(Wqkv,  wqkv16,  nWqkv);
  cvt_f32_to_f16_kernel<<<2048, 256, 0, stream>>>(Wproj, wproj16, nWproj);
  cvt_f32_to_f16_kernel<<<2048, 256, 0, stream>>>(W1,    w116,    nW1);
  cvt_f32_to_f16_kernel<<<2048, 256, 0, stream>>>(W2,    w216,    nW2);

  const float* xin = x;
  for (int l = 0; l < Ll; ++l) {
    const _Float16* wq = wqkv16  + (size_t)l * C3 * Cc;
    const _Float16* wp = wproj16 + (size_t)l * Cc * Cc;
    const _Float16* wa = w116    + (size_t)l * MLPn * Cc;
    const _Float16* wb = w216    + (size_t)l * Cc * MLPn;

    // 1) LN1 -> h16
    ln_kernel<<<M, 256, 0, stream>>>(xin, ln1g + (size_t)l * Cc, ln1b + (size_t)l * Cc, h16, Cc);
    // 2) QKV GEMM -> qkv16 (f16)
    gemm_wmma_kernel<<<dim3(C3 / 64, M / 128), 128, 0, stream>>>(
        h16, wq, bqkv + (size_t)l * C3, nullptr, nullptr, qkv16, M, C3, Cc, GEMM_F16);
    // 3) attention -> o16
    attn_kernel<<<dim3(Nn / 64, Hh, Bb), 128, 0, stream>>>(qkv16, o16, Nn, C3, Cc, scale);
    // 4) proj GEMM + residual -> bufM (f32)
    gemm_wmma_kernel<<<dim3(Cc / 64, M / 128), 128, 0, stream>>>(
        o16, wp, bproj + (size_t)l * Cc, xin, bufM, nullptr, M, Cc, Cc, GEMM_F32_RES);
    // 5) LN2 -> h16
    ln_kernel<<<M, 256, 0, stream>>>(bufM, ln2g + (size_t)l * Cc, ln2b + (size_t)l * Cc, h16, Cc);
    // 6) MLP up + GELU -> ff16
    gemm_wmma_kernel<<<dim3(MLPn / 64, M / 128), 128, 0, stream>>>(
        h16, wa, b1 + (size_t)l * MLPn, nullptr, nullptr, ff16, M, MLPn, Cc, GEMM_F16_GELU);
    // 7) MLP down + residual -> next x (last layer writes d_out)
    float* xout = (l == Ll - 1) ? (float*)d_out : ((l & 1) ? bufB : bufA);
    gemm_wmma_kernel<<<dim3(Cc / 64, M / 128), 128, 0, stream>>>(
        ff16, wb, b2 + (size_t)l * Cc, bufM, xout, nullptr, M, Cc, MLPn, GEMM_F32_RES);
    xin = xout;
  }
}
